// EnetGnn_69810398429304
// MI455X (gfx1250) — matrix operand order repaired and checked by reading, not compiled, not run on hardware
//
#include <hip/hip_runtime.h>
#include <hip/hip_bf16.h>

// ---------------------------------------------------------------------------
// EnetGnn for MI455X (gfx1250, wave32, WMMA).
//
// Dominant cost: r = X X^T per batch (34.4 GFLOP fp32) fused with top-16-
// smallest selection.  Done with split-bf16 (hi+lo) v_wmma_f32_16x16x32_bf16:
// r ~= hi*hi + hi*lo + lo*hi  (lo*lo ~ 2^-16 relative, dropped).
// Remaining ~3 GFLOP stages run in plain fp32 VALU (precision-sensitive
// softmax logits), bandwidth-trivial vs 23.3 TB/s HBM.
// ---------------------------------------------------------------------------

typedef __bf16 bf16_t;
typedef __attribute__((ext_vector_type(8)))  __bf16 v8bf;
typedef __attribute__((ext_vector_type(16))) __bf16 v16bf;
typedef __attribute__((ext_vector_type(8)))  float  v8f;

#define NB   4
#define CDIM 256
#define C2   128
#define HWSZ 4096
#define KNN  16

// ---------------------------------------------------------------------------
// Stage P: transpose cat [n][c][i] -> X [n][i][c]; emit fp32 copy and
// split-bf16 pair (hi = rne(x), lo = rne(x - hi)).
// ---------------------------------------------------------------------------
__global__ __launch_bounds__(256) void k_preprocess(const float* __restrict__ cat,
                                                    bf16_t* __restrict__ Xhi,
                                                    bf16_t* __restrict__ Xlo,
                                                    float*  __restrict__ Xf32) {
    long t = (long)blockIdx.x * blockDim.x + threadIdx.x;   // == ((n*256+c)*4096+i)
    if (t >= (long)NB * CDIM * HWSZ) return;
    int  i  = (int)(t & (HWSZ - 1));
    long nc = t >> 12;
    int  c  = (int)(nc & (CDIM - 1));
    int  n  = (int)(nc >> 8);
    float x = cat[t];
    long o = ((long)(n * HWSZ + i)) * CDIM + c;
    Xf32[o] = x;
    bf16_t hi = (bf16_t)x;
    Xhi[o] = hi;
    Xlo[o] = (bf16_t)(x - (float)hi);
}

// ---------------------------------------------------------------------------
// Stage A: fused distance GEMM + streaming top-16-smallest per row.
// grid = N * (HW/16) blocks; 256 threads = 8 waves.
// Each wave: 32 j-tiles x 8 K-steps x 3 WMMAs, then insertion-select into
// per-wave sorted lists in LDS.  Final per-row 8-way merge.
// ---------------------------------------------------------------------------
__global__ __launch_bounds__(256) void k_knn(const bf16_t* __restrict__ Xhi,
                                             const bf16_t* __restrict__ Xlo,
                                             int* __restrict__ idx_out) {
    const int n    = blockIdx.x >> 8;
    const int it   = blockIdx.x & 255;
    const int tid  = threadIdx.x;
    const int w    = tid >> 5;
    const int lane = tid & 31;

    __shared__ __align__(32) bf16_t Ahi[16 * CDIM];
    __shared__ __align__(32) bf16_t Alo[16 * CDIM];
    __shared__ float stage[8][16][16];
    __shared__ float lval[8][16][16];
    __shared__ int   lidx[8][16][16];
    __shared__ float fval[16][16];
    __shared__ int   fidx[16][16];

    const bf16_t* Xhi_n = Xhi + (long)n * HWSZ * CDIM;
    const bf16_t* Xlo_n = Xlo + (long)n * HWSZ * CDIM;

    // cooperative A-tile load (rows it*16..it*16+15, all K)
    for (int e = tid; e < 16 * CDIM; e += 256) {
        int r = e >> 8;
        int c = e & (CDIM - 1);
        long src = (long)(it * 16 + r) * CDIM + c;
        Ahi[e] = Xhi_n[src];
        Alo[e] = Xlo_n[src];
    }
    // init per-wave top-16 lists (ascending sorted; [15] is current max)
    for (int e = tid; e < 8 * 16 * 16; e += 256) {
        ((float*)lval)[e] = 3.0e38f;
        ((int*)lidx)[e]   = 0;
    }
    __syncthreads();

    const int m     = lane & 15;        // A row / B col / C col for this lane
    const int khalf = lane >> 4;        // 0: low K half, 1: high K half

    for (int s = 0; s < 32; ++s) {
        const int jt    = w * 32 + s;   // each wave owns 32 of the 256 j-tiles
        const int jbase = jt * 16;
        v8f acc = {};

        #pragma unroll
        for (int ks = 0; ks < 8; ++ks) {
            const int kb = ks * 32;
            // --- A fragment (16x32 bf16): lane holds
            //     row m, K = {kb+16h..kb+16h+15} (contiguous, matches B)
            const bf16_t* pah = &Ahi[m * CDIM + kb + 16 * khalf];
            const bf16_t* pal = &Alo[m * CDIM + kb + 16 * khalf];
            v8bf ah0 = *(const v8bf*)pah;
            v8bf ah1 = *(const v8bf*)(pah + 8);
            v8bf al0 = *(const v8bf*)pal;
            v8bf al1 = *(const v8bf*)(pal + 8);
            v16bf ah = __builtin_shufflevector(ah0, ah1, 0,1,2,3,4,5,6,7,8,9,10,11,12,13,14,15);
            v16bf al = __builtin_shufflevector(al0, al1, 0,1,2,3,4,5,6,7,8,9,10,11,12,13,14,15);
            // --- B fragment (32x16 bf16): lane holds column m (= row j of X),
            //     16 contiguous K starting at kb + 16*khalf -> one 32B load.
            long boff = (long)(jbase + m) * CDIM + kb + 16 * khalf;
            v16bf bh = *(const v16bf*)&Xhi_n[boff];
            v16bf bl = *(const v16bf*)&Xlo_n[boff];
            // split-bf16 product: hi*hi + hi*lo + lo*hi
            acc = __builtin_amdgcn_wmma_f32_16x16x32_bf16(false, ah, false, bh, (short)0, acc, false, false);
            acc = __builtin_amdgcn_wmma_f32_16x16x32_bf16(false, ah, false, bl, (short)0, acc, false, false);
            acc = __builtin_amdgcn_wmma_f32_16x16x32_bf16(false, al, false, bh, (short)0, acc, false, false);
        }

        // C layout: VGPR v -> row v + 8*khalf, col m
        #pragma unroll
        for (int v = 0; v < 8; ++v)
            stage[w][v + 8 * khalf][m] = acc[v];
        asm volatile("s_wait_dscnt 0x0" ::: "memory");   // cross-lane LDS RAW

        // streaming selection: 16 lanes, one row each (per-wave lists -> no races)
        if (lane < 16) {
            const int row = lane;
            #pragma unroll 1
            for (int col = 0; col < 16; ++col) {
                float vv = stage[w][row][col];
                if (vv < lval[w][row][15]) {
                    int p = 15;
                    while (p > 0 && lval[w][row][p - 1] > vv) {
                        lval[w][row][p] = lval[w][row][p - 1];
                        lidx[w][row][p] = lidx[w][row][p - 1];
                        --p;
                    }
                    lval[w][row][p] = vv;
                    lidx[w][row][p] = jbase + col;
                }
            }
        }
    }
    __syncthreads();

    // merge the 8 per-wave sorted lists into the global 16-smallest per row
    if (tid < 16) {
        const int row = tid;
        for (int e = 0; e < 16; ++e) {
            fval[row][e] = lval[0][row][e];
            fidx[row][e] = lidx[0][row][e];
        }
        for (int ww = 1; ww < 8; ++ww) {
            for (int e = 0; e < 16; ++e) {
                float vv = lval[ww][row][e];
                if (vv < fval[row][15]) {
                    int p = 15;
                    while (p > 0 && fval[row][p - 1] > vv) {
                        fval[row][p] = fval[row][p - 1];
                        fidx[row][p] = fidx[row][p - 1];
                        --p;
                    }
                    fval[row][p] = vv;
                    fidx[row][p] = lidx[ww][row][e];
                } else {
                    break;  // source list ascending -> rest can't qualify
                }
            }
        }
        int* out = idx_out + ((long)(n * HWSZ) + it * 16 + row) * KNN;
        for (int e = 0; e < 16; ++e) out[e] = fidx[row][e];
    }
}

// ---------------------------------------------------------------------------
// Stage B: hm[rid][d] = mean_16(X0[nbr]) . W_g[d] + b_g[d]
// (mean<->linear interchange; neighbors come from BATCH 0 rows — the
//  reference's faithful local-index quirk.)
// ---------------------------------------------------------------------------
__global__ __launch_bounds__(128) void k_hm(const float* __restrict__ Xf32,
                                            const int* __restrict__ idx,
                                            const float* __restrict__ Wg,
                                            const float* __restrict__ bg,
                                            float* __restrict__ hm) {
    const int rid = blockIdx.x;      // n*HW + i
    const int t   = threadIdx.x;     // 0..127
    __shared__ float hmean[CDIM];
    __shared__ int   nb[KNN];
    if (t < KNN) nb[t] = idx[(long)rid * KNN + t];
    __syncthreads();
    for (int c = t; c < CDIM; c += 128) {
        float s = 0.0f;
        #pragma unroll
        for (int e = 0; e < KNN; ++e)
            s += Xf32[(long)nb[e] * CDIM + c];   // batch-0 rows only (faithful)
        hmean[c] = s * (1.0f / (float)KNN);
    }
    __syncthreads();
    float accv = bg[t];
    const float* wrow = Wg + (long)t * CDIM;
    for (int c = 0; c < CDIM; ++c)
        accv = fmaf(hmean[c], wrow[c], accv);
    hm[(long)rid * C2 + t] = accv;
}

// ---------------------------------------------------------------------------
// Stage C: att[n][c][d] = softmax_d( sum_i hm[n][i][c]*hm[n][i][d] )
// one block per (n,c) logits row; block-wide max/sum reduction.
// ---------------------------------------------------------------------------
__global__ __launch_bounds__(128) void k_att(const float* __restrict__ hm,
                                             float* __restrict__ att) {
    const int n = blockIdx.x >> 7;
    const int c = blockIdx.x & 127;
    const int d = threadIdx.x;
    const float* H = hm + (long)n * HWSZ * C2;
    float s = 0.0f;
    for (int i = 0; i < HWSZ; ++i)
        s = fmaf(H[(long)i * C2 + c], H[(long)i * C2 + d], s);

    __shared__ float red[C2];
    red[d] = s;
    __syncthreads();
    for (int off = 64; off > 0; off >>= 1) {
        if (d < off) red[d] = fmaxf(red[d], red[d + off]);
        __syncthreads();
    }
    float mx = red[0];
    __syncthreads();
    float e = __expf(s - mx);
    red[d] = e;
    __syncthreads();
    for (int off = 64; off > 0; off >>= 1) {
        if (d < off) red[d] += red[d + off];
        __syncthreads();
    }
    float sum = red[0];
    att[((long)n * C2 + c) * C2 + d] = e / sum;
}

// ---------------------------------------------------------------------------
// Stage D: out[n][d][i] = gamma * sum_c rgb_in[n][c][i]*att[n][c][d]
//                         + rgb_in[n][d][i]
// att cached in LDS; two 64-wide register accumulator passes per thread.
// ---------------------------------------------------------------------------
__global__ __launch_bounds__(256) void k_out(const float* __restrict__ rgb_in,
                                             const float* __restrict__ att,
                                             const float* __restrict__ gamma,
                                             float* __restrict__ out) {
    const int n  = blockIdx.x >> 4;
    const int ib = blockIdx.x & 15;
    const int t  = threadIdx.x;
    const int i  = ib * 256 + t;
    __shared__ float As[C2 * C2];
    for (int e = t; e < C2 * C2; e += 256)
        As[e] = att[(long)n * C2 * C2 + e];
    __syncthreads();

    const float* rin = rgb_in + (long)n * C2 * HWSZ;
    const float  g   = gamma[0];
    for (int half = 0; half < 2; ++half) {
        float acc[64];
        #pragma unroll
        for (int d = 0; d < 64; ++d) acc[d] = 0.0f;
        for (int c = 0; c < C2; ++c) {
            float x = rin[(long)c * HWSZ + i];
            const float* arow = &As[c * C2 + half * 64];
            #pragma unroll
            for (int d = 0; d < 64; ++d)
                acc[d] = fmaf(x, arow[d], acc[d]);
        }
        #pragma unroll
        for (int d = 0; d < 64; ++d) {
            int dd = half * 64 + d;
            out[((long)n * C2 + dd) * HWSZ + i] =
                fmaf(g, acc[d], rin[(long)dd * HWSZ + i]);
        }
    }
}

// ---------------------------------------------------------------------------
extern "C" void kernel_launch(void* const* d_in, const int* in_sizes, int n_in,
                              void* d_out, int out_size, void* d_ws, size_t ws_size,
                              hipStream_t stream) {
    const float* cat    = (const float*)d_in[0];
    const float* rgb_in = (const float*)d_in[1];
    const float* Wg     = (const float*)d_in[2];
    const float* bg     = (const float*)d_in[3];
    const float* gamma  = (const float*)d_in[4];
    // d_in[5]=gnn_iterations(==1), d_in[6]=k(==16): fixed by the reference setup.

    char* ws = (char*)d_ws;
    bf16_t* Xhi  = (bf16_t*)(ws);                       //  8 MB
    bf16_t* Xlo  = (bf16_t*)(ws + (size_t)( 8u << 20)); //  8 MB
    float*  Xf32 = (float*) (ws + (size_t)(16u << 20)); // 16 MB
    int*    idx  = (int*)   (ws + (size_t)(32u << 20)); //  1 MB
    float*  hm   = (float*) (ws + (size_t)(33u << 20)); //  8 MB
    float*  att  = (float*) (ws + (size_t)(41u << 20)); //  0.25 MB

    k_preprocess<<<(NB * CDIM * HWSZ) / 256, 256, 0, stream>>>(cat, Xhi, Xlo, Xf32);
    k_knn<<<NB * (HWSZ / 16), 256, 0, stream>>>(Xhi, Xlo, idx);
    k_hm<<<NB * HWSZ, 128, 0, stream>>>(Xf32, idx, Wg, bg, hm);
    k_att<<<NB * C2, 128, 0, stream>>>(hm, att);
    k_out<<<NB * (HWSZ / 256), 256, 0, stream>>>(rgb_in, att, gamma, (float*)d_out);
}
